// MultiHGNNCritic_9216999817278
// MI455X (gfx1250) — matrix operand order repaired and checked by reading, not compiled, not run on hardware
//
#include <hip/hip_runtime.h>
#include <stdint.h>

typedef __attribute__((ext_vector_type(16))) __bf16 v16bf;
typedef __attribute__((ext_vector_type(8)))  __bf16 v8bf;
typedef __attribute__((ext_vector_type(8)))  float  v8f;
typedef __attribute__((ext_vector_type(4)))  float  v4f;

#define BM 128
#define BN 128
#define BK 32
#define KPAD 40   // padded LDS row (bf16 elems): rows start 80B apart, 16B chunks stay aligned

// ---------- CDNA5 async global->LDS copy (ASYNCcnt path) ----------
__device__ __forceinline__ void async_copy16(uint32_t lds_off, const __bf16* gptr) {
  asm volatile("global_load_async_to_lds_b128 %0, %1, off"
               :: "v"(lds_off), "v"((uint64_t)(uintptr_t)gptr)
               : "memory");
}
__device__ __forceinline__ void wait_async0() {
  asm volatile("s_wait_asynccnt 0x0" ::: "memory");
}

// ---------------- one-time conversion / transpose kernels ----------------

__global__ __launch_bounds__(256) void cvt_bf16_kernel(const float* __restrict__ in,
                                                       __bf16* __restrict__ out) {
  size_t i = ((size_t)blockIdx.x * 256 + threadIdx.x) * 4;
  v4f v = *(const v4f*)(in + i);
  out[i + 0] = (__bf16)v[0]; out[i + 1] = (__bf16)v[1];
  out[i + 2] = (__bf16)v[2]; out[i + 3] = (__bf16)v[3];
}

// out[C][R] = (bf16) in[R][C]   (32x32 LDS tiles; R,C multiples of 32)
__global__ __launch_bounds__(256) void transpose_cvt_kernel(const float* __restrict__ in,
                                                            __bf16* __restrict__ out,
                                                            int R, int C) {
  __shared__ float tile[32][33];
  int bx = blockIdx.x * 32, by = blockIdx.y * 32;
  int tx = threadIdx.x & 31, ty = threadIdx.x >> 5;  // 32x8
  for (int r = ty; r < 32; r += 8)
    tile[r][tx] = in[(size_t)(by + r) * C + bx + tx];
  __syncthreads();
  for (int r = ty; r < 32; r += 8)
    out[(size_t)(bx + r) * R + by + tx] = (__bf16)tile[tx][r];
}

// ---------------- degree kernels (read original f32 H) ----------------

__global__ __launch_bounds__(256) void dv_isqrt_kernel(const float* __restrict__ H,
                                                       float* __restrict__ dv, int E) {
  int idx = blockIdx.x * 256 + threadIdx.x;          // over T*N
  const v4f* r4 = (const v4f*)(H + (size_t)idx * E);
  float s = 0.f;
  for (int i = 0; i < E / 4; ++i) { v4f v = r4[i]; s += v[0] + v[1] + v[2] + v[3]; }
  dv[idx] = rsqrtf(s);
}

__global__ __launch_bounds__(256) void de_inv_kernel(const float* __restrict__ H,
                                                     float* __restrict__ de, int N, int E) {
  int idx = blockIdx.x * 256 + threadIdx.x;          // over T*E
  int t = idx / E, e = idx % E;
  const float* base = H + (size_t)t * N * E + e;
  float s = 0.f;
  for (int n = 0; n < N; ++n) s += base[(size_t)n * E];
  de[idx] = 1.0f / s;
}

// ---------------- bf16 WMMA GEMM, async-staged, double-buffered LDS ----------------
// C[M,Nc] = epilogue( A @ B ),  epilogue: v += rowbias[m]; v *= colscale[n]; relu
//   A  : bf16 row-major [M,K], lda
//   B  : if FAST_B: Bt = B^T row-major [Nc,K], ldb = row stride of Bt
//        else:      B row-major [K,Nc],       ldb = row stride of B (transposed LDS staging)
//   C  : bf16 row-major [M,Nc], ldc
template <bool FAST_B>
__global__ __launch_bounds__(256) void gemm_bf16_wmma(
    const __bf16* __restrict__ A, const __bf16* __restrict__ B, __bf16* __restrict__ C,
    int M, int Nc, int K, int lda, int ldb, int ldc,
    const float* __restrict__ rowbias, const float* __restrict__ colscale, int do_relu) {
  __shared__ __align__(16) __bf16 ldsA[2][BM * KPAD];   // [m][k]
  __shared__ __align__(16) __bf16 ldsB[2][BN * KPAD];   // [n][k]

  const int tid  = threadIdx.x;
  const int lane = tid & 31;
  const int wave = tid >> 5;
  const int waveM = wave & 3;   // 4 waves along M
  const int waveN = wave >> 2;  // 2 waves along N
  const int bm0 = blockIdx.y * BM;
  const int bn0 = blockIdx.x * BN;

  const uint32_t ldsA0 = (uint32_t)(uintptr_t)&ldsA[0][0];
  const uint32_t ldsA1 = (uint32_t)(uintptr_t)&ldsA[1][0];
  const uint32_t ldsB0 = (uint32_t)(uintptr_t)&ldsB[0][0];
  const uint32_t ldsB1 = (uint32_t)(uintptr_t)&ldsB[1][0];

  // stage one k-tile into LDS buffer `buf` (async 16B chunks; B fallback is u16 copies)
  auto stage = [&](int buf, int k0) {
    uint32_t aoff = buf ? ldsA1 : ldsA0;
    // A tile: 128 rows x 64B = 512 chunks; 2 per thread
#pragma unroll
    for (int c = tid; c < 512; c += 256) {
      int row = c >> 2, sub = c & 3;
      async_copy16(aoff + row * (KPAD * 2) + sub * 16,
                   A + (size_t)(bm0 + row) * lda + k0 + sub * 8);
    }
    if (FAST_B) {
      uint32_t boff = buf ? ldsB1 : ldsB0;
#pragma unroll
      for (int c = tid; c < 512; c += 256) {
        int row = c >> 2, sub = c & 3;
        async_copy16(boff + row * (KPAD * 2) + sub * 16,
                     B + (size_t)(bn0 + row) * ldb + k0 + sub * 8);
      }
    } else {
      __bf16* bp = &ldsB[buf][0];
      for (int i = tid; i < BK * BN; i += 256) {   // coalesced along n
        int k = i >> 7, n = i & 127;
        bp[n * KPAD + k] = B[(size_t)(k0 + k) * ldb + bn0 + n];
      }
    }
  };

  const v8f zero = {0.f, 0.f, 0.f, 0.f, 0.f, 0.f, 0.f, 0.f};
  v8f acc[2][4];
#pragma unroll
  for (int i = 0; i < 2; ++i)
#pragma unroll
    for (int j = 0; j < 4; ++j) acc[i][j] = zero;

  const int mrow = lane & 15;
  const int hi16 = lane >> 4;
  const int ako  = hi16 * 8;    // A: K chunks {0..7}/{8..15}, then +16
  const int bko  = hi16 * 8;    // B: K chunks {0..7}/{8..15}, then +16

  const int ktiles = K / BK;
  stage(0, 0);
  wait_async0();
  __syncthreads();

  for (int kt = 0; kt < ktiles; ++kt) {
    int cur = kt & 1;
    if (kt + 1 < ktiles) stage(cur ^ 1, (kt + 1) * BK);   // prefetch next tile

    const __bf16* la = &ldsA[cur][0];
    const __bf16* lb = &ldsB[cur][0];
    v16bf afrag[2], bfrag[4];
#pragma unroll
    for (int i = 0; i < 2; ++i) {
      int m = waveM * 32 + i * 16 + mrow;
      v8bf lo = *(const v8bf*)(la + m * KPAD + ako);
      v8bf hi = *(const v8bf*)(la + m * KPAD + ako + 16);
      afrag[i] = __builtin_shufflevector(lo, hi, 0, 1, 2, 3, 4, 5, 6, 7,
                                         8, 9, 10, 11, 12, 13, 14, 15);
    }
#pragma unroll
    for (int j = 0; j < 4; ++j) {
      int n = waveN * 64 + j * 16 + mrow;
      v8bf lo = *(const v8bf*)(lb + n * KPAD + bko);
      v8bf hi = *(const v8bf*)(lb + n * KPAD + bko + 16);
      bfrag[j] = __builtin_shufflevector(lo, hi, 0, 1, 2, 3, 4, 5, 6, 7,
                                         8, 9, 10, 11, 12, 13, 14, 15);
    }
#pragma unroll
    for (int i = 0; i < 2; ++i)
#pragma unroll
      for (int j = 0; j < 4; ++j)
        acc[i][j] = __builtin_amdgcn_wmma_f32_16x16x32_bf16(
            false, afrag[i], false, bfrag[j], (short)0, acc[i][j], false, false);

    wait_async0();
    __syncthreads();
  }

  // epilogue + bf16 store (C layout: VGPR r -> M = r + hi16*8, N = lane&15)
#pragma unroll
  for (int i = 0; i < 2; ++i) {
    int rbase = bm0 + waveM * 32 + i * 16 + hi16 * 8;
#pragma unroll
    for (int j = 0; j < 4; ++j) {
      int colg = bn0 + waveN * 64 + j * 16 + mrow;
      float cs = colscale ? colscale[colg] : 1.f;
#pragma unroll
      for (int r = 0; r < 8; ++r) {
        int rowg = rbase + r;
        float v = acc[i][j][r];
        if (rowbias) v += rowbias[rowg];
        v *= cs;
        if (do_relu) v = fmaxf(v, 0.f);
        C[(size_t)rowg * ldc + colg] = (__bf16)v;
      }
    }
  }
}

// ---------------- vals[n] = sum_h hT[h][n]*Wv[h] + bv  (coalesced over n) ----------------
__global__ __launch_bounds__(256) void matvecT_kernel(const __bf16* __restrict__ hT,
                                                      const float* __restrict__ w,
                                                      const float* __restrict__ bvp,
                                                      float* __restrict__ out, int D, int Nc) {
  int n = blockIdx.x * 256 + threadIdx.x;
  float s = 0.f;
  for (int h = 0; h < D; ++h) s += (float)hT[(size_t)h * Nc + n] * w[h];
  out[n] = s + bvp[0];
}

// ---------------- out = vals . Wm + bm ----------------
__global__ __launch_bounds__(256) void final_kernel(const float* __restrict__ vals,
                                                    const float* __restrict__ Wm,
                                                    const float* __restrict__ bm,
                                                    float* __restrict__ out, int n) {
  __shared__ float red[256];
  float s = 0.f;
  for (int i = threadIdx.x; i < n; i += 256) s += vals[i] * Wm[i];
  red[threadIdx.x] = s;
  __syncthreads();
  for (int st = 128; st; st >>= 1) {
    if (threadIdx.x < st) red[threadIdx.x] += red[threadIdx.x + st];
    __syncthreads();
  }
  if (threadIdx.x == 0) out[0] = red[0] + bm[0];
}

extern "C" void kernel_launch(void* const* d_in, const int* in_sizes, int n_in,
                              void* d_out, int out_size, void* d_ws, size_t ws_size,
                              hipStream_t stream) {
  (void)in_sizes; (void)n_in; (void)out_size; (void)ws_size;
  const float* X  = (const float*)d_in[0];
  const float* H  = (const float*)d_in[1];
  const float* W0 = (const float*)d_in[2];
  const float* b0 = (const float*)d_in[3];
  const float* W1 = (const float*)d_in[4];
  const float* b1 = (const float*)d_in[5];
  const float* Wv = (const float*)d_in[6];
  const float* bv = (const float*)d_in[7];
  const float* Wm = (const float*)d_in[8];
  const float* bm = (const float*)d_in[9];
  float* out = (float*)d_out;

  constexpr int T = 4, Nn = 4096, E = 1024, OBS = 512, HID = 1024;

  // ---- workspace layout ----
  float* dv   = (float*)d_ws;               // T*Nn
  float* de   = dv + T * Nn;                // T*E
  float* vals = de + T * E;                 // T*Nn
  __bf16* b   = (__bf16*)(vals + T * Nn);
  __bf16* Xbf  = b; b += (size_t)Nn * OBS;          // X        [n][o]
  __bf16* Hbf  = b; b += (size_t)T * Nn * E;        // H        [t][n][e]
  __bf16* HbfT = b; b += (size_t)T * Nn * E;        // H^T      [t][e][n]
  __bf16* W0T  = b; b += (size_t)T * HID * OBS;     // W0^T     [t][h][o]
  __bf16* W1T  = b; b += (size_t)T * HID * HID;     // W1^T     [t][h'][h]
  __bf16* C1   = b; b += (size_t)HID * Nn;          // Xn^T     [h][n]
  __bf16* XeT  = b; b += (size_t)HID * E;           // Xe^T     [h][e]
  __bf16* hT   = b; b += (size_t)HID * Nn;          // h^T      [h][n]

  // ---- one-time bf16 conversions / transposes ----
  cvt_bf16_kernel<<<(Nn * OBS) / 1024, 256, 0, stream>>>(X, Xbf);
  cvt_bf16_kernel<<<((size_t)T * Nn * E) / 1024, 256, 0, stream>>>(H, Hbf);
  for (int t = 0; t < T; ++t) {
    transpose_cvt_kernel<<<dim3(E / 32, Nn / 32), 256, 0, stream>>>(
        H + (size_t)t * Nn * E, HbfT + (size_t)t * Nn * E, Nn, E);
    transpose_cvt_kernel<<<dim3(HID / 32, OBS / 32), 256, 0, stream>>>(
        W0 + (size_t)t * OBS * HID, W0T + (size_t)t * HID * OBS, OBS, HID);
    transpose_cvt_kernel<<<dim3(HID / 32, HID / 32), 256, 0, stream>>>(
        W1 + (size_t)t * HID * HID, W1T + (size_t)t * HID * HID, HID, HID);
  }
  dv_isqrt_kernel<<<T * Nn / 256, 256, 0, stream>>>(H, dv, E);
  de_inv_kernel<<<T * E / 256, 256, 0, stream>>>(H, de, Nn, E);

  for (int t = 0; t < T; ++t) {
    const __bf16* Ht  = Hbf + (size_t)t * Nn * E;
    const __bf16* HtT = HbfT + (size_t)t * Nn * E;
    const float* dvt = dv + t * Nn;
    const float* det = de + t * E;

    // 1: C1 = Xn0^T = (W0^T @ X^T + b0[m]) * dv[n]
    gemm_bf16_wmma<true><<<dim3(Nn / BN, HID / BM), 256, 0, stream>>>(
        W0T + (size_t)t * HID * OBS, Xbf, C1, HID, Nn, OBS, OBS, OBS, Nn,
        b0 + t * HID, dvt, 0);
    // 2: XeT = (C1 @ H) * de[n]      (B^T = HbfT, row-major)
    gemm_bf16_wmma<true><<<dim3(E / BN, HID / BM), 256, 0, stream>>>(
        C1, HtT, XeT, HID, E, Nn, Nn, Nn, E, nullptr, det, 0);
    // 3: hT = relu((XeT @ H^T) * dv[n])   (B^T = Hbf, row-major)
    gemm_bf16_wmma<true><<<dim3(Nn / BN, HID / BM), 256, 0, stream>>>(
        XeT, Ht, hT, HID, Nn, E, E, E, Nn, nullptr, dvt, 1);
    // 4: C1 = Xn1^T = (W1^T @ h1^T + b1[m]) * dv[n]   (B = hT row-major [K][Nc] -> slow staging)
    gemm_bf16_wmma<false><<<dim3(Nn / BN, HID / BM), 256, 0, stream>>>(
        W1T + (size_t)t * HID * HID, hT, C1, HID, Nn, HID, HID, Nn, Nn,
        b1 + t * HID, dvt, 0);
    // 5: XeT = (C1 @ H) * de[n]
    gemm_bf16_wmma<true><<<dim3(E / BN, HID / BM), 256, 0, stream>>>(
        C1, HtT, XeT, HID, E, Nn, Nn, Nn, E, nullptr, det, 0);
    // 6: hT = relu((XeT @ H^T) * dv[n])
    gemm_bf16_wmma<true><<<dim3(Nn / BN, HID / BM), 256, 0, stream>>>(
        XeT, Ht, hT, HID, Nn, E, E, E, Nn, nullptr, dvt, 1);
    // 7: vals[t] = h3 @ Wv + bv   (reads hT coalesced over n)
    matvecT_kernel<<<Nn / 256, 256, 0, stream>>>(hT, Wv + t * HID, bv + t,
                                                 vals + t * Nn, HID, Nn);
  }

  final_kernel<<<1, 256, 0, stream>>>(vals, Wm, bm, out, T * Nn);
}
